// DeAltHDR_7035156430885
// MI455X (gfx1250) — matrix operand order repaired and checked by reading, not compiled, not run on hardware
//
#include <hip/hip_runtime.h>
#include <hip/hip_bf16.h>

typedef __attribute__((ext_vector_type(16))) __bf16 v16bf;
typedef __attribute__((ext_vector_type(8)))  __bf16 v8bf;
typedef __attribute__((ext_vector_type(8)))  float  v8f;
typedef __attribute__((ext_vector_type(4)))  float  v4f;

// ======================= elementwise / small kernels =======================

__global__ void pack_bf16_kernel(const float* __restrict__ src, __bf16* __restrict__ dst, int n) {
    int i = blockIdx.x * blockDim.x + threadIdx.x;
    if (i < n) dst[i] = (__bf16)src[i];
}

// transpose-pack: (C,N) f32 -> (N,C) bf16, 32x32 LDS tiles, per batch
__global__ void transpose_pack_kernel(const float* __restrict__ src, __bf16* __restrict__ dst,
                                      int Cc, int Nn) {
    __shared__ __bf16 tile[32][33];
    int nb = blockIdx.x * 32;     // pixel base
    int cb = blockIdx.y * 32;     // channel base
    int b  = blockIdx.z;
    const float* sp = src + (size_t)b * Cc * Nn;
    __bf16* dp = dst + (size_t)b * Cc * Nn;
    int tx = threadIdx.x;         // 32
    int ty = threadIdx.y;         // 8
#pragma unroll
    for (int j = 0; j < 4; ++j) {
        int c = cb + ty + j * 8;
        tile[ty + j * 8][tx] = (__bf16)sp[(size_t)c * Nn + nb + tx];
    }
    __syncthreads();
#pragma unroll
    for (int j = 0; j < 4; ++j) {
        int n = nb + ty + j * 8;
        dp[(size_t)n * Cc + cb + tx] = tile[tx][ty + j * 8];
    }
}

// dst[b][0:3]=a[b], dst[b][3:6]=bimg[b]; chN = 3*H*W
__global__ void concat2_kernel(const float* __restrict__ a, const float* __restrict__ bimg,
                               float* __restrict__ dst, int total, int chN) {
    int i = blockIdx.x * blockDim.x + threadIdx.x;
    if (i >= total) return;
    int r  = i % (2 * chN);
    int bb = i / (2 * chN);
    dst[i] = (r < chN) ? a[(size_t)bb * chN + r] : bimg[(size_t)bb * chN + (r - chN)];
}

// generic direct conv (spynet only: tiny)
__global__ void conv2d_kernel(const float* __restrict__ in, const float* __restrict__ w,
                              const float* __restrict__ bias, float* __restrict__ out,
                              int total, int Cin, int Cout, int Hin, int Win,
                              int Hout, int Wout, int ks, int stride, int pad, int relu) {
    int idx = blockIdx.x * blockDim.x + threadIdx.x;
    if (idx >= total) return;
    int x  = idx % Wout;
    int y  = (idx / Wout) % Hout;
    int co = (idx / (Wout * Hout)) % Cout;
    int b  = idx / (Wout * Hout * Cout);
    float acc = bias ? bias[co] : 0.0f;
    for (int ci = 0; ci < Cin; ++ci) {
        const float* ip = in + ((size_t)b * Cin + ci) * Hin * Win;
        const float* wp = w + ((size_t)co * Cin + ci) * ks * ks;
        for (int ky = 0; ky < ks; ++ky) {
            int iy = y * stride - pad + ky;
            if (iy < 0 || iy >= Hin) continue;
            for (int kx = 0; kx < ks; ++kx) {
                int ix = x * stride - pad + kx;
                if (ix < 0 || ix >= Win) continue;
                acc += ip[(size_t)iy * Win + ix] * wp[ky * ks + kx];
            }
        }
    }
    if (relu) acc = fmaxf(acc, 0.0f);
    out[idx] = acc;
}

// bilinear resize (half-pixel centers), NCHW
__global__ void resize_bilinear_kernel(const float* __restrict__ in, float* __restrict__ out,
                                       int total, int Ch, int Hin, int Win, int Hout, int Wout) {
    int idx = blockIdx.x * blockDim.x + threadIdx.x;
    if (idx >= total) return;
    int x = idx % Wout;
    int y = (idx / Wout) % Hout;
    int bc = idx / (Wout * Hout); // b*Ch + c
    float sx = ((float)x + 0.5f) * (float)Win / (float)Wout - 0.5f;
    float sy = ((float)y + 0.5f) * (float)Hin / (float)Hout - 0.5f;
    sx = fminf(fmaxf(sx, 0.0f), (float)(Win - 1));
    sy = fminf(fmaxf(sy, 0.0f), (float)(Hin - 1));
    int x0 = (int)floorf(sx), y0 = (int)floorf(sy);
    int x1 = min(x0 + 1, Win - 1), y1 = min(y0 + 1, Hin - 1);
    float wx = sx - (float)x0, wy = sy - (float)y0;
    const float* ip = in + (size_t)bc * Hin * Win;
    float v = ip[(size_t)y0 * Win + x0] * (1.f - wx) * (1.f - wy)
            + ip[(size_t)y0 * Win + x1] * wx * (1.f - wy)
            + ip[(size_t)y1 * Win + x0] * (1.f - wx) * wy
            + ip[(size_t)y1 * Win + x1] * wx * wy;
    out[idx] = v;
}

// flow warp, exact reference math; dst has its own batch stride so we can
// scatter straight into the 385-channel output.
__global__ void warp_kernel(const float* __restrict__ x, size_t x_bstride,
                            const float* __restrict__ flow,
                            float* __restrict__ dst, size_t dst_bstride,
                            int total, int C, int H, int W) {
    int idx = blockIdx.x * blockDim.x + threadIdx.x;
    if (idx >= total) return;
    int px = idx % W;
    int py = (idx / W) % H;
    int c  = (idx / (W * H)) % C;
    int b  = idx / (W * H * C);
    int N = H * W;
    float fx = flow[((size_t)b * 2 + 0) * N + py * W + px];
    float fy = flow[((size_t)b * 2 + 1) * N + py * W + px];
    float vx = (float)px + fx;
    float vy = (float)py + fy;
    float ix = vx * (float)W / (float)(W - 1) - 0.5f;
    float iy = vy * (float)H / (float)(H - 1) - 0.5f;
    ix = fminf(fmaxf(ix, 0.0f), (float)(W - 1));
    iy = fminf(fmaxf(iy, 0.0f), (float)(H - 1));
    float x0f = floorf(ix), y0f = floorf(iy);
    float wx = ix - x0f, wy = iy - y0f;
    int x0 = (int)x0f, y0 = (int)y0f;
    int x1 = min(x0 + 1, W - 1), y1 = min(y0 + 1, H - 1);
    const float* ip = x + (size_t)b * x_bstride + (size_t)c * N;
    float v = ip[(size_t)y0 * W + x0] * (1.f - wx) * (1.f - wy)
            + ip[(size_t)y0 * W + x1] * wx * (1.f - wy)
            + ip[(size_t)y1 * W + x0] * (1.f - wx) * wy
            + ip[(size_t)y1 * W + x1] * wx * wy;
    dst[(size_t)b * dst_bstride + (size_t)c * N + py * W + px] = v;
}

// mean |wb - cur| over 3 channels, threshold -> mask (ws copy + output ch 192)
__global__ void mask_kernel(const float* __restrict__ wb, const float* __restrict__ cur,
                            float* __restrict__ maskb, float* __restrict__ out,
                            int total, int N, int outC) {
    int idx = blockIdx.x * blockDim.x + threadIdx.x;
    if (idx >= total) return;
    int n = idx % N;
    int b = idx / N;
    float d = 0.f;
    for (int c = 0; c < 3; ++c) {
        size_t o = ((size_t)b * 3 + c) * N + n;
        d += fabsf(wb[o] - cur[o]);
    }
    d *= (1.0f / 3.0f);
    float m = (d > (0.5f * 255.0f / 300.0f)) ? 1.0f : 0.0f;
    maskb[idx] = m;
    out[((size_t)b * outC + 192) * N + n] = m;
}

// depthwise 3x3 pad 1, bf16 in/out; weight channel offset selects qkv slice
__global__ void dwconv3x3_kernel(const __bf16* __restrict__ in, const float* __restrict__ w,
                                 __bf16* __restrict__ out, int total, int rows,
                                 int H, int W, int wch_off) {
    int idx = blockIdx.x * blockDim.x + threadIdx.x;
    if (idx >= total) return;
    int x = idx % W;
    int y = (idx / W) % H;
    int c = (idx / (W * H)) % rows;
    int b = idx / (W * H * rows);
    const __bf16* ip = in + ((size_t)b * rows + c) * H * W;
    const float* wp = w + (size_t)(wch_off + c) * 9;
    float acc = 0.f;
    for (int ky = 0; ky < 3; ++ky) {
        int iy = y - 1 + ky;
        if (iy < 0 || iy >= H) continue;
        for (int kx = 0; kx < 3; ++kx) {
            int ix = x - 1 + kx;
            if (ix < 0 || ix >= W) continue;
            acc += (float)ip[(size_t)iy * W + ix] * wp[ky * 3 + kx];
        }
    }
    out[idx] = (__bf16)acc;
}

// per-row sum of squares (optionally masked), one block per (b,row)
__global__ void rowsumsq_kernel(const __bf16* __restrict__ X, size_t batch_stride, int rows,
                                const float* __restrict__ mask, float* __restrict__ out, int N) {
    int row = blockIdx.x;
    int b = row / rows, r = row % rows;
    const __bf16* xp = X + (size_t)b * batch_stride + (size_t)r * N;
    const float* mp = mask ? (mask + (size_t)b * N) : nullptr;
    float s = 0.f;
    for (int i = threadIdx.x; i < N; i += blockDim.x) {
        float v = (float)xp[i];
        if (mp) v *= mp[i];
        s += v * v;
    }
    __shared__ float red[256];
    red[threadIdx.x] = s;
    __syncthreads();
    for (int off = 128; off > 0; off >>= 1) {
        if ((int)threadIdx.x < off) red[threadIdx.x] += red[threadIdx.x + off];
        __syncthreads();
    }
    if (threadIdx.x == 0) out[row] = red[0];
}

// in-place: X = X * mask / max(sqrt(ss), 1e-12)
__global__ void scale_rows_kernel(__bf16* __restrict__ X, size_t batch_stride, int rows,
                                  const float* __restrict__ mask, const float* __restrict__ ss,
                                  int total, int N) {
    int idx = blockIdx.x * blockDim.x + threadIdx.x;
    if (idx >= total) return;
    int n = idx % N;
    int r = (idx / N) % rows;
    int b = idx / (N * rows);
    float nr = fmaxf(sqrtf(ss[b * rows + r]), 1e-12f);
    size_t o = (size_t)b * batch_stride + (size_t)r * N + n;
    float v = (float)X[o];
    if (mask) v *= mask[(size_t)b * N + n];
    X[o] = (__bf16)(v / nr);
}

// softmax over 48 entries, emit bf16 row padded to 64 with zeros
__global__ void softmax48_kernel(const float* __restrict__ attn, __bf16* __restrict__ out) {
    int row = blockIdx.x;                 // B*4*48 rows
    const float* a = attn + (size_t)row * 48;
    __bf16* o = out + (size_t)row * 64;
    int t = threadIdx.x;                  // 64 threads
    float v = (t < 48) ? a[t] : -__builtin_inff();
    __shared__ float red[64];
    red[t] = v;
    __syncthreads();
    for (int off = 32; off > 0; off >>= 1) {
        if (t < off) red[t] = fmaxf(red[t], red[t + off]);
        __syncthreads();
    }
    float mx = red[0];
    __syncthreads();
    float e = (t < 48) ? __expf(v - mx) : 0.0f;
    red[t] = e;
    __syncthreads();
    for (int off = 32; off > 0; off >>= 1) {
        if (t < off) red[t] += red[t + off];
        __syncthreads();
    }
    float s = red[0];
    o[t] = (__bf16)((t < 48) ? (e / s) : 0.0f);
}

// ======================= WMMA GEMM kernels =======================
// Fragment layout used here (wave32, contiguous k order for A and B):
//  A 16x32 bf16: lane<16 -> M=lane, a[0..7]=K(kb..kb+7),  a[8..15]=K(kb+8..kb+15)
//                lane>=16 -> M=lane-16, a[0..7]=K(kb+16..), a[8..15]=K(kb+24..)
//  B 32x16 bf16: lane<16 -> col=lane, b[i]=B[kb+i][col]; lane>=16 -> K += 16
//  C/D 16x16 f32: lane<16 -> col=lane, d[i]=D[m=i][col]; lane>=16 -> m=i+8

__device__ __forceinline__ v16bf make_frag(v8bf lo, v8bf hi) {
    v16bf f;
#pragma unroll
    for (int i = 0; i < 8; ++i) { f[i] = lo[i]; f[i + 8] = hi[i]; }
    return f;
}

__device__ __forceinline__ v8f wmma_bf16(v16bf a, v16bf b, v8f c) {
    return __builtin_amdgcn_wmma_f32_16x16x32_bf16(false, a, false, b, (short)0, c, false, false);
}

// Transposed-formulation GEMM: out(m,p) = sum_k W(m,k) * XT(p,k).
// A operand = XT (N x K, streaming, contiguous k-runs).
// B operand = W columns == contiguous rows of row-major W (M x K).
// D rows are pixels (fast axis of (M,N) buffers) -> contiguous vector stores.
// Each wave: one 16-pixel tile x two 16-channel tiles (A fragment shared).
template <int N, int K, bool OUT_BF16>
__global__ void gemm_wmma_t_kernel(const __bf16* __restrict__ XT, size_t x_bstride,
                                   const __bf16* __restrict__ Wt,
                                   void* __restrict__ Dptr, size_t d_bstride, int d_row_off) {
    int lane = threadIdx.x & 31;
    int tile_p = (blockIdx.x * blockDim.y + threadIdx.y) * 16;  // pixel tile
    int tile_c = blockIdx.y * 32;                               // 2 channel tiles
    int b = blockIdx.z;
    int half = lane >> 4;
    int l15  = lane & 15;
    const __bf16* arow  = XT + (size_t)b * x_bstride + (size_t)(tile_p + l15) * K + half * 16;
    const __bf16* brow0 = Wt + (size_t)(tile_c + l15) * K + half * 16;
    const __bf16* brow1 = brow0 + (size_t)16 * K;
    v8f acc0 = {}, acc1 = {};
#pragma unroll
    for (int kb = 0; kb < K; kb += 32) {
        v16bf afrag = make_frag(*(const v8bf*)(arow + kb), *(const v8bf*)(arow + kb + 8));
        v16bf bf0   = make_frag(*(const v8bf*)(brow0 + kb), *(const v8bf*)(brow0 + kb + 8));
        v16bf bf1   = make_frag(*(const v8bf*)(brow1 + kb), *(const v8bf*)(brow1 + kb + 8));
        acc0 = wmma_bf16(afrag, bf0, acc0);
        acc1 = wmma_bf16(afrag, bf1, acc1);
    }
    int p0 = tile_p + half * 8;
    int c0 = tile_c + l15;
    int c1 = c0 + 16;
    if constexpr (OUT_BF16) {
        __bf16* dp = (__bf16*)Dptr + (size_t)b * d_bstride;
        v8bf o0, o1;
#pragma unroll
        for (int i = 0; i < 8; ++i) { o0[i] = (__bf16)acc0[i]; o1[i] = (__bf16)acc1[i]; }
        *(v8bf*)(dp + (size_t)(d_row_off + c0) * N + p0) = o0;
        *(v8bf*)(dp + (size_t)(d_row_off + c1) * N + p0) = o1;
    } else {
        float* dp = (float*)Dptr + (size_t)b * d_bstride;
        v4f o0a = {acc0[0], acc0[1], acc0[2], acc0[3]};
        v4f o0b = {acc0[4], acc0[5], acc0[6], acc0[7]};
        v4f o1a = {acc1[0], acc1[1], acc1[2], acc1[3]};
        v4f o1b = {acc1[4], acc1[5], acc1[6], acc1[7]};
        *(v4f*)(dp + (size_t)(d_row_off + c0) * N + p0)     = o0a;
        *(v4f*)(dp + (size_t)(d_row_off + c0) * N + p0 + 4) = o0b;
        *(v4f*)(dp + (size_t)(d_row_off + c1) * N + p0)     = o1a;
        *(v4f*)(dp + (size_t)(d_row_off + c1) * N + p0 + 4) = o1b;
    }
}

// attn(c,d) = sum_n q(c,n)*k(d,n) * temp[h]; K-dim = N. Both operands are
// contiguous row streams (B operand is k^T); pure pointer-bump loop.
template <int N>
__global__ void attn_qk_kernel(const __bf16* __restrict__ Qn, const __bf16* __restrict__ Kv,
                               const float* __restrict__ temp, float* __restrict__ attn,
                               int C, int kvrows, int hd) {
    int lane = threadIdx.x & 31;
    int tile_m = blockIdx.y * 16;
    int tile_n = blockIdx.x * 16;
    int bh = blockIdx.z;
    int b = bh >> 2, h = bh & 3;
    int half = lane >> 4;
    int l15  = lane & 15;
    const __bf16* qrow = Qn + ((size_t)b * C + (size_t)h * hd + tile_m + l15) * N + half * 16;
    const __bf16* krow = Kv + ((size_t)b * kvrows + (size_t)h * hd + tile_n + l15) * N + half * 16;
    v8f acc = {};
    for (int nb = 0; nb < N; nb += 32) {
        v16bf afrag = make_frag(*(const v8bf*)qrow, *(const v8bf*)(qrow + 8));
        v16bf bfrag = make_frag(*(const v8bf*)krow, *(const v8bf*)(krow + 8));
        __builtin_prefetch(qrow + 1024, 0, 1);
        qrow += 32;
        krow += 32;
        acc = wmma_bf16(afrag, bfrag, acc);
    }
    float t = temp[h];
    int om = tile_m + half * 8;
    int on = tile_n + l15;
    float* ap = attn + (size_t)bh * hd * hd;
#pragma unroll
    for (int i = 0; i < 8; ++i) ap[(size_t)(om + i) * hd + on] = acc[i] * t;
}

// out(c,p) = sum_d attn_sm(c,d) * v(d,p); K padded 48->64. Output written
// TRANSPOSED into attT (B,N,C) so stores are contiguous v8bf and the proj
// GEMM consumes a contiguous A stream.
template <int N>
__global__ void attn_v_kernel(const __bf16* __restrict__ Asm, const __bf16* __restrict__ Kv,
                              __bf16* __restrict__ attT, int C, int kvrows, int hd) {
    int lane = threadIdx.x & 31;
    int tile_m = blockIdx.y * 16;
    int tile_n = blockIdx.x * 16;
    int bh = blockIdx.z;
    int b = bh >> 2, h = bh & 3;
    int half = lane >> 4;
    int l15  = lane & 15;
    const __bf16* arow = Asm + ((size_t)bh * hd + tile_m + l15) * 64 + half * 16;
    const __bf16* vcol = Kv + ((size_t)b * kvrows + (size_t)(C + h * hd) + (size_t)half * 16) * N
                       + (tile_n + l15);
    v8f acc = {};
    // kb = 0: rows 0..31 all valid
    {
        v16bf afrag = make_frag(*(const v8bf*)arow, *(const v8bf*)(arow + 8));
        v16bf bfrag;
#pragma unroll
        for (int i = 0; i < 16; ++i) bfrag[i] = vcol[(size_t)i * N];
        acc = wmma_bf16(afrag, bfrag, acc);
    }
    // kb = 32: half 0 -> rows 32..47 valid; half 1 -> rows 48..63 are zero pad
    {
        v16bf afrag = make_frag(*(const v8bf*)(arow + 32), *(const v8bf*)(arow + 40));
        const __bf16* vcol2 = vcol + (size_t)32 * N;
        v16bf bfrag;
#pragma unroll
        for (int i = 0; i < 16; ++i)
            bfrag[i] = half ? (__bf16)0.0f : vcol2[(size_t)i * N];
        acc = wmma_bf16(afrag, bfrag, acc);
    }
    int om = tile_m + half * 8;     // channel within head
    int on = tile_n + l15;          // pixel
    v8bf ov;
#pragma unroll
    for (int i = 0; i < 8; ++i) ov[i] = (__bf16)acc[i];
    *(v8bf*)(attT + ((size_t)b * N + on) * C + h * hd + om) = ov;
}

// ======================= host orchestration =======================

static inline int cdiv(int a, int b) { return (a + b - 1) / b; }

extern "C" void kernel_launch(void* const* d_in, const int* in_sizes, int n_in,
                              void* d_out, int out_size, void* d_ws, size_t ws_size,
                              hipStream_t stream) {
    (void)in_sizes; (void)n_in; (void)out_size; (void)ws_size;
    const float* cur_feat = (const float*)d_in[0];
    const float* ref_feat = (const float*)d_in[1];
    const float* cur_img  = (const float*)d_in[2];
    const float* ref_img  = (const float*)d_in[3];
    const float* sp_w[6]  = {(const float*)d_in[4],  (const float*)d_in[6],
                             (const float*)d_in[8],  (const float*)d_in[10],
                             (const float*)d_in[12], (const float*)d_in[14]};
    const float* sp_b[6]  = {(const float*)d_in[5],  (const float*)d_in[7],
                             (const float*)d_in[9],  (const float*)d_in[11],
                             (const float*)d_in[13], (const float*)d_in[15]};
    const float* qkv_w  = (const float*)d_in[16];
    const float* dw_w   = (const float*)d_in[17];
    const float* proj_w = (const float*)d_in[18];
    const float* temp   = (const float*)d_in[19];
    float* out = (float*)d_out;

    constexpr int B = 2, C = 192, H = 192, W = 192, N = H * W, HD = 48, OUTC = 2 * C + 1;

    // ---- workspace bump allocator ----
    char* ws = (char*)d_ws;
    size_t off = 0;
    auto alloc = [&](size_t bytes) -> char* {
        char* p = ws + off;
        off += (bytes + 255) & ~(size_t)255;
        return p;
    };
    float*  xcat   = (float*)alloc((size_t)B * 6 * N * 4);
    float*  pa     = (float*)alloc((size_t)B * 32 * 96 * 96 * 4);
    float*  pb     = (float*)alloc((size_t)B * 64 * 48 * 48 * 4);
    float*  flowF  = (float*)alloc((size_t)B * 2 * N * 4);
    float*  flowB  = (float*)alloc((size_t)B * 2 * N * 4);
    float*  imgw   = (float*)alloc((size_t)B * 3 * N * 4);
    float*  wback  = (float*)alloc((size_t)B * 3 * N * 4);
    float*  maskb  = (float*)alloc((size_t)B * N * 4);
    __bf16* wq_bf  = (__bf16*)alloc((size_t)576 * 192 * 2);
    __bf16* wp_bf  = (__bf16*)alloc((size_t)192 * 192 * 2);
    __bf16* featT  = (__bf16*)alloc((size_t)B * N * C * 2);     // (B,N,C) transposed features
    __bf16* prebf  = (__bf16*)alloc((size_t)B * 384 * N * 2);   // 192 (cur) or 384 (ref) rows
    __bf16* qbuf   = (__bf16*)alloc((size_t)B * 192 * N * 2);
    __bf16* kvbuf  = (__bf16*)alloc((size_t)B * 384 * N * 2);   // rows 0..191=k, 192..383=v
    __bf16* attT   = (__bf16*)alloc((size_t)B * N * C * 2);     // (B,N,C) attention output
    float*  attnf  = (float*)alloc((size_t)B * 4 * HD * HD * 4);
    __bf16* attnsm = (__bf16*)alloc((size_t)B * 4 * HD * 64 * 2);
    float*  normq  = (float*)alloc((size_t)B * 192 * 4);
    float*  normk  = (float*)alloc((size_t)B * 192 * 4);

    const int TB = 256;
    auto g1 = [&](int total) { return dim3(cdiv(total, TB)); };

    // ---- spynet (both directions) ----
    struct Lyr { int cin, cout, hin, hout, ks, st, pad; };
    const Lyr L[6] = {{6, 32, 192, 96, 7, 2, 3},  {32, 64, 96, 48, 7, 2, 3},
                      {64, 96, 48, 24, 5, 2, 2},  {96, 128, 24, 12, 3, 2, 1},
                      {128, 192, 12, 6, 3, 2, 1}, {192, 2, 6, 6, 3, 1, 1}};
    auto run_spynet = [&](const float* i1, const float* i2, float* flow_out) {
        int tc = B * 6 * N;
        concat2_kernel<<<g1(tc), TB, 0, stream>>>(i1, i2, xcat, tc, 3 * N);
        const float* src = xcat;
        float* bufs[2] = {pa, pb};
        for (int l = 0; l < 6; ++l) {
            float* dst = bufs[l & 1];
            int total = B * L[l].cout * L[l].hout * L[l].hout;
            conv2d_kernel<<<g1(total), TB, 0, stream>>>(
                src, sp_w[l], sp_b[l], dst, total,
                L[l].cin, L[l].cout, L[l].hin, L[l].hin,
                L[l].hout, L[l].hout, L[l].ks, L[l].st, L[l].pad, (l < 5) ? 1 : 0);
            src = dst;
        }
        int tr = B * 2 * N;   // resize 6x6 -> 192x192 (flow values NOT rescaled, per reference)
        resize_bilinear_kernel<<<g1(tr), TB, 0, stream>>>(src, flow_out, tr, 2, 6, 6, H, W);
    };
    run_spynet(cur_img, ref_img, flowF);
    run_spynet(ref_img, cur_img, flowB);

    // ---- warps + mask ----
    int t3 = B * 3 * N;
    warp_kernel<<<g1(t3), TB, 0, stream>>>(cur_img, (size_t)3 * N, flowF,
                                           imgw, (size_t)3 * N, t3, 3, H, W);
    warp_kernel<<<g1(t3), TB, 0, stream>>>(imgw, (size_t)3 * N, flowB,
                                           wback, (size_t)3 * N, t3, 3, H, W);
    int tm = B * N;
    mask_kernel<<<g1(tm), TB, 0, stream>>>(wback, cur_img, maskb, out, tm, N, OUTC);
    // warped_feat straight into output channels 0..191
    int tf = B * C * N;
    warp_kernel<<<g1(tf), TB, 0, stream>>>(ref_feat, (size_t)C * N, flowB,
                                           out, (size_t)OUTC * N, tf, C, H, W);

    // ---- pack weights ----
    pack_bf16_kernel<<<g1(576 * 192), TB, 0, stream>>>(qkv_w, wq_bf, 576 * 192);
    pack_bf16_kernel<<<g1(192 * 192), TB, 0, stream>>>(proj_w, wp_bf, 192 * 192);

    dim3 trBlk(32, 8, 1);
    dim3 trGrid(N / 32, C / 32, B);
    dim3 gemmBlk(32, 4, 1);
    // ---- current: only q rows (0..191) of qkv needed ----
    transpose_pack_kernel<<<trGrid, trBlk, 0, stream>>>(cur_feat, featT, C, N);
    gemm_wmma_t_kernel<N, C, true><<<dim3(N / 64, 192 / 32, B), gemmBlk, 0, stream>>>(
        featT, (size_t)N * C, wq_bf, prebf, (size_t)192 * N, 0);
    int tq = B * 192 * N;
    dwconv3x3_kernel<<<g1(tq), TB, 0, stream>>>(prebf, dw_w, qbuf, tq, 192, H, W, 0);

    // ---- ref: k,v rows (192..575) ----
    transpose_pack_kernel<<<trGrid, trBlk, 0, stream>>>(ref_feat, featT, C, N);
    gemm_wmma_t_kernel<N, C, true><<<dim3(N / 64, 384 / 32, B), gemmBlk, 0, stream>>>(
        featT, (size_t)N * C, wq_bf + (size_t)192 * C, prebf, (size_t)384 * N, 0);
    int tkv = B * 384 * N;
    dwconv3x3_kernel<<<g1(tkv), TB, 0, stream>>>(prebf, dw_w, kvbuf, tkv, 384, H, W, 192);

    // ---- L2 norms (q masked first), in-place normalize ----
    rowsumsq_kernel<<<dim3(B * 192), 256, 0, stream>>>(qbuf, (size_t)192 * N, 192,
                                                       maskb, normq, N);
    rowsumsq_kernel<<<dim3(B * 192), 256, 0, stream>>>(kvbuf, (size_t)384 * N, 192,
                                                       (const float*)nullptr, normk, N);
    scale_rows_kernel<<<g1(tq), TB, 0, stream>>>(qbuf, (size_t)192 * N, 192,
                                                 maskb, normq, tq, N);
    scale_rows_kernel<<<g1(tq), TB, 0, stream>>>(kvbuf, (size_t)384 * N, 192,
                                                 (const float*)nullptr, normk, tq, N);

    // ---- attention ----
    attn_qk_kernel<N><<<dim3(HD / 16, HD / 16, B * 4), 32, 0, stream>>>(
        qbuf, kvbuf, temp, attnf, C, 384, HD);
    softmax48_kernel<<<dim3(B * 4 * HD), 64, 0, stream>>>(attnf, attnsm);
    attn_v_kernel<N><<<dim3(N / 16, HD / 16, B * 4), 32, 0, stream>>>(
        attnsm, kvbuf, attT, C, 384, HD);

    // ---- proj GEMM (transposed form) -> output channels 193..384 (f32) ----
    gemm_wmma_t_kernel<N, C, false><<<dim3(N / 64, 192 / 32, B), gemmBlk, 0, stream>>>(
        attT, (size_t)N * C, wp_bf, out, (size_t)OUTC * N, 193);
}
